// GlobBlock_4217657884931
// MI455X (gfx1250) — matrix operand order repaired and checked by reading, not compiled, not run on hardware
//
#include <hip/hip_runtime.h>
#include <hip/hip_bf16.h>
#include <math.h>

typedef __attribute__((ext_vector_type(8)))  __bf16 v8bf;
typedef __attribute__((ext_vector_type(16))) __bf16 v16bf;
typedef __attribute__((ext_vector_type(8)))  float  v8f;

#define NNODES  200000
#define NGRAPHS 2000
#define WID     512
#define HID     1024

__device__ __forceinline__ float gelu_exact(float x) {
    return 0.5f * x * (1.0f + erff(x * 0.70710678118654752f));
}

// ---------------------------------------------------------------------------
// Weight convert + transpose:  in f32 [K, C]  ->  out bf16 [C, K]
// ---------------------------------------------------------------------------
__global__ __launch_bounds__(256) void convert_transpose_kernel(
    const float* __restrict__ in, __bf16* __restrict__ out, int K, int C) {
    size_t idx = (size_t)blockIdx.x * 256 + threadIdx.x;
    if (idx >= (size_t)K * C) return;
    int k = (int)(idx / C);
    int c = (int)(idx % C);
    out[(size_t)c * K + k] = (__bf16)in[idx];
}

__global__ __launch_bounds__(256) void zero_f32_kernel(float* __restrict__ p, int n) {
    int i = blockIdx.x * 256 + threadIdx.x;
    if (i < n) p[i] = 0.0f;
}

__global__ __launch_bounds__(256) void f32_to_bf16_kernel(
    const float* __restrict__ in, __bf16* __restrict__ out, int n) {
    int i = blockIdx.x * 256 + threadIdx.x;
    if (i < n) out[i] = (__bf16)in[i];
}

// ---------------------------------------------------------------------------
// Segment sum of xx (bf16 [N,512]) into pooled (f32 [G,512]).
// batch is sorted, so run-length accumulate locally, flush with f32 atomics
// (pooled is 4 MB -> L2-resident atomics).
// Thread g: col = g%512, node chunk = g/512 covering 128 nodes.
// ---------------------------------------------------------------------------
__global__ __launch_bounds__(256) void segsum_kernel(
    const __bf16* __restrict__ xx, const int* __restrict__ batch,
    float* __restrict__ pooled, int nnodes) {
    int g    = blockIdx.x * 256 + threadIdx.x;
    int col  = g & 511;
    int n0   = (g >> 9) * 128;
    if (n0 >= nnodes) return;
    int n1 = n0 + 128; if (n1 > nnodes) n1 = nnodes;
    int cur = batch[n0];
    float run = 0.0f;
    for (int n = n0; n < n1; ++n) {
        int b = batch[n];
        if (b != cur) {
            atomicAdd(&pooled[(size_t)cur * WID + col], run);
            run = 0.0f; cur = b;
        }
        run += (float)xx[(size_t)n * WID + col];
    }
    atomicAdd(&pooled[(size_t)cur * WID + col], run);
}

// ---------------------------------------------------------------------------
// Fused WMMA GEMM:  D[16 rows x COLS] = A[16 x K] * Bt^T  (+ epilogue)
//   A  : bf16 [rows, K] (or f32 if AF32)      row-major activations
//   Bt : bf16 [COLS, K]                       pre-transposed weights
// 8 waves / block; wave w owns cols [w*NTILES*16, ...); block owns 16 rows
// and the ENTIRE output width (needed so EPI=0 can LayerNorm a full row).
// EPI: 0 = +bias, LayerNorm, store bf16        (pre)
//      1 = gelu(+bias) + exp(z)*gather(msg), store bf16   (mem)
//      2 = gelu(+bias), store f32              (msg)
//      3 = +bias -> out2 ; x + out2 -> out1    (post)
// ---------------------------------------------------------------------------
template<int NTILES, int EPI, bool AF32>
__global__ __launch_bounds__(256) void wmma_gemm(
    const void*  __restrict__ Av,
    const __bf16* __restrict__ Bt,
    const float* __restrict__ bias,
    __bf16*      __restrict__ outB,
    float*       __restrict__ outF0,
    float*       __restrict__ outF1,
    const float* __restrict__ gatherSrc,
    const int*   __restrict__ batch,
    const float* __restrict__ addSrc,
    const float* __restrict__ zscale,
    int K)
{
    constexpr int COLS = NTILES * 16 * 8;   // 8 waves
    __shared__ float s_sum[16];
    __shared__ float s_sq[16];
    __shared__ int   s_batch[16];

    const int tid     = threadIdx.x;
    const int wave    = tid >> 5;
    const int lane    = tid & 31;
    const int lm      = lane & 15;          // M (A) / N (B,C/D) sub-index
    const int half    = lane >> 4;          // lane half selects K sub-range
    const int rowBase = blockIdx.x * 16;
    const int wcol    = wave * (NTILES * 16);

    if (EPI == 0) { if (tid < 16) { s_sum[tid] = 0.0f; s_sq[tid] = 0.0f; } }
    if (EPI == 1) { if (tid < 16) { s_batch[tid] = batch[rowBase + tid]; } }
    if (EPI == 0 || EPI == 1) __syncthreads();

    // A-fragment base: row = rowBase+lm, K offset half*8 (16-bit A layout)
    const __bf16* aB = nullptr;
    const float*  aF = nullptr;
    if (AF32) aF = (const float*) Av + (size_t)(rowBase + lm) * K + half * 8;
    else      aB = (const __bf16*)Av + (size_t)(rowBase + lm) * K + half * 8;

    // B-fragment bases: col n = wcol+t*16+lm, K offset half*8 (16-bit B layout)
    const __bf16* bp[NTILES];
#pragma unroll
    for (int t = 0; t < NTILES; ++t)
        bp[t] = Bt + (size_t)(wcol + t * 16 + lm) * K + half * 8;

    v8f acc[NTILES];
#pragma unroll
    for (int t = 0; t < NTILES; ++t)
#pragma unroll
        for (int i = 0; i < 8; ++i) acc[t][i] = 0.0f;

    for (int kk = 0; kk < K; kk += 32) {
        v16bf afrag;
        if (AF32) {
            v8f lo = *(const v8f*)(aF + kk);
            v8f hi = *(const v8f*)(aF + kk + 16);
#pragma unroll
            for (int i = 0; i < 8; ++i) { afrag[i] = (__bf16)lo[i]; afrag[i + 8] = (__bf16)hi[i]; }
        } else {
            v8bf lo = *(const v8bf*)(aB + kk);
            v8bf hi = *(const v8bf*)(aB + kk + 16);
            afrag = __builtin_shufflevector(lo, hi, 0,1,2,3,4,5,6,7,8,9,10,11,12,13,14,15);
        }
#pragma unroll
        for (int t = 0; t < NTILES; ++t) {
            v8bf blo = *(const v8bf*)(bp[t] + kk);
            v8bf bhi = *(const v8bf*)(bp[t] + kk + 16);
            v16bf bfrag = __builtin_shufflevector(blo, bhi, 0,1,2,3,4,5,6,7,8,9,10,11,12,13,14,15);
            acc[t] = __builtin_amdgcn_wmma_f32_16x16x32_bf16(
                false, afrag, false, bfrag, (short)0, acc[t], false, false);
        }
    }

    // C/D layout: VGPR i, lanes 0-15 -> row i, lanes 16-31 -> row i+8; col = lane%16
    if (EPI == 0) {
        float rs[8], rq[8];
#pragma unroll
        for (int i = 0; i < 8; ++i) { rs[i] = 0.0f; rq[i] = 0.0f; }
#pragma unroll
        for (int t = 0; t < NTILES; ++t) {
            float b = bias[wcol + t * 16 + lm];
#pragma unroll
            for (int i = 0; i < 8; ++i) {
                float v = acc[t][i] + b; acc[t][i] = v;
                rs[i] += v; rq[i] += v * v;
            }
        }
#pragma unroll
        for (int m = 1; m < 16; m <<= 1)
#pragma unroll
            for (int i = 0; i < 8; ++i) {
                rs[i] += __shfl_xor(rs[i], m, 32);
                rq[i] += __shfl_xor(rq[i], m, 32);
            }
        if (lm == 0)
#pragma unroll
            for (int i = 0; i < 8; ++i) {
                atomicAdd(&s_sum[i + half * 8], rs[i]);
                atomicAdd(&s_sq [i + half * 8], rq[i]);
            }
        __syncthreads();
#pragma unroll
        for (int i = 0; i < 8; ++i) {
            int r = i + half * 8;
            float mu   = s_sum[r] * (1.0f / (float)COLS);
            float var  = s_sq[r]  * (1.0f / (float)COLS) - mu * mu;
            float rstd = rsqrtf(var + 1e-5f);
            size_t row = (size_t)(rowBase + r);
#pragma unroll
            for (int t = 0; t < NTILES; ++t) {
                int col = wcol + t * 16 + lm;
                outB[row * COLS + col] = (__bf16)((acc[t][i] - mu) * rstd);
            }
        }
    } else if (EPI == 1) {
        float sc = expf(zscale[0]);
#pragma unroll
        for (int t = 0; t < NTILES; ++t) {
            int col = wcol + t * 16 + lm;
            float b = bias[col];
#pragma unroll
            for (int i = 0; i < 8; ++i) {
                int r = i + half * 8;
                float v = gelu_exact(acc[t][i] + b)
                        + sc * gatherSrc[(size_t)s_batch[r] * COLS + col];
                outB[(size_t)(rowBase + r) * COLS + col] = (__bf16)v;
            }
        }
    } else if (EPI == 2) {
#pragma unroll
        for (int t = 0; t < NTILES; ++t) {
            int col = wcol + t * 16 + lm;
            float b = bias[col];
#pragma unroll
            for (int i = 0; i < 8; ++i) {
                int r = i + half * 8;
                outF0[(size_t)(rowBase + r) * COLS + col] = gelu_exact(acc[t][i] + b);
            }
        }
    } else { // EPI == 3
#pragma unroll
        for (int t = 0; t < NTILES; ++t) {
            int col = wcol + t * 16 + lm;
            float b = bias[col];
#pragma unroll
            for (int i = 0; i < 8; ++i) {
                int r = i + half * 8;
                size_t idx = (size_t)(rowBase + r) * COLS + col;
                float o2 = acc[t][i] + b;
                outF1[idx] = o2;
                outF0[idx] = addSrc[idx] + o2;
            }
        }
    }
}

// ---------------------------------------------------------------------------
extern "C" void kernel_launch(void* const* d_in, const int* in_sizes, int n_in,
                              void* d_out, int out_size, void* d_ws, size_t ws_size,
                              hipStream_t stream) {
    (void)in_sizes; (void)n_in; (void)out_size; (void)ws_size;
    const float* x      = (const float*)d_in[0];
    const int*   batch  = (const int*)  d_in[1];
    const float* pre_w  = (const float*)d_in[2];
    const float* pre_b  = (const float*)d_in[3];
    const float* mem_w  = (const float*)d_in[4];
    const float* mem_b  = (const float*)d_in[5];
    const float* msg_w  = (const float*)d_in[6];
    const float* msg_b  = (const float*)d_in[7];
    const float* post_w = (const float*)d_in[8];
    const float* post_b = (const float*)d_in[9];
    const float* zero   = (const float*)d_in[10];

    char* ws = (char*)d_ws;
    size_t off = 0;
    auto alloc = [&](size_t bytes) -> char* {
        char* p = ws + off;
        off += (bytes + 255) & ~(size_t)255;
        return p;
    };
    __bf16* xx      = (__bf16*)alloc((size_t)NNODES  * WID * 2);
    __bf16* tbuf    = (__bf16*)alloc((size_t)NNODES  * HID * 2);
    float*  pooled  = (float*) alloc((size_t)NGRAPHS * WID * 4);
    __bf16* pooledB = (__bf16*)alloc((size_t)NGRAPHS * WID * 2);
    float*  msg     = (float*) alloc((size_t)NGRAPHS * HID * 4);
    __bf16* preWt   = (__bf16*)alloc((size_t)WID * WID * 2);
    __bf16* memWt   = (__bf16*)alloc((size_t)WID * HID * 2);
    __bf16* msgWt   = (__bf16*)alloc((size_t)WID * HID * 2);
    __bf16* postWt  = (__bf16*)alloc((size_t)HID * WID * 2);

    // Weight prep (bf16, transposed to [out,K]) + pooled zero-init
    convert_transpose_kernel<<<(WID*WID+255)/256, 256, 0, stream>>>(pre_w,  preWt,  WID, WID);
    convert_transpose_kernel<<<(WID*HID+255)/256, 256, 0, stream>>>(mem_w,  memWt,  WID, HID);
    convert_transpose_kernel<<<(WID*HID+255)/256, 256, 0, stream>>>(msg_w,  msgWt,  WID, HID);
    convert_transpose_kernel<<<(HID*WID+255)/256, 256, 0, stream>>>(post_w, postWt, HID, WID);
    zero_f32_kernel<<<(NGRAPHS*WID+255)/256, 256, 0, stream>>>(pooled, NGRAPHS*WID);

    // pre: xx = LayerNorm(x @ pre_w + pre_b)        [N,512] bf16
    wmma_gemm<4, 0, true><<<NNODES/16, 256, 0, stream>>>(
        (const void*)x, preWt, pre_b, xx, nullptr, nullptr,
        nullptr, nullptr, nullptr, nullptr, WID);

    // pooled = segment_sum(xx, batch)
    int chunks = (NNODES + 127) / 128;
    segsum_kernel<<<(chunks * WID) / 256, 256, 0, stream>>>(xx, batch, pooled, NNODES);
    f32_to_bf16_kernel<<<(NGRAPHS*WID+255)/256, 256, 0, stream>>>(pooled, pooledB, NGRAPHS*WID);

    // msg = gelu(pooled @ msg_w + msg_b)            [G,1024] f32
    wmma_gemm<8, 2, false><<<NGRAPHS/16, 256, 0, stream>>>(
        (const void*)pooledB, msgWt, msg_b, nullptr, msg, nullptr,
        nullptr, nullptr, nullptr, nullptr, WID);

    // t = gelu(xx @ mem_w + mem_b) + exp(z)*msg[batch]   [N,1024] bf16
    wmma_gemm<8, 1, false><<<NNODES/16, 256, 0, stream>>>(
        (const void*)xx, memWt, mem_b, tbuf, nullptr, nullptr,
        msg, batch, nullptr, zero, WID);

    // out2 = t @ post_w + post_b ; out1 = x + out2
    float* out1 = (float*)d_out;
    float* out2 = out1 + (size_t)NNODES * WID;
    wmma_gemm<4, 3, false><<<NNODES/16, 256, 0, stream>>>(
        (const void*)tbuf, postWt, post_b, nullptr, out1, out2,
        nullptr, nullptr, x, nullptr, HID);
}
